// Block_25718264168914
// MI455X (gfx1250) — matrix operand order repaired and checked by reading, not compiled, not run on hardware
//
#include <hip/hip_runtime.h>
#include <math.h>

#define C_    256
#define NH_   8
#define HD_   32
#define NT_   49     // tokens per window
#define NP_   64     // padded tokens
#define SCALE_ 0.17677669529663687f  // 32^-0.5

// padded LDS row strides: row-to-row bank shift = 4 banks (16B) -> 16 lanes
// of a b128 fragment load cover all 64 banks conflict-free
#define SH_STR 264   // bf16 rows of 256 (+8 pad)   528B/row
#define SV_STR 72    // bf16 rows of 64  (+8 pad)   144B/row
#define AF_STR 68    // f32  rows of 64  (+4 pad)   272B/row
#define AB_STR 72    // bf16 rows of 64  (+8 pad)   144B/row
#define SM_STR 1032  // bf16 rows of 1024 (+8 pad)  2064B/row

typedef unsigned short u16;
typedef __attribute__((ext_vector_type(16))) __bf16 v16bf;
typedef __attribute__((ext_vector_type(8)))  float  v8f;

__device__ __forceinline__ u16 f2bf(float f) {
    union { float f; unsigned u; } x; x.f = f;
    unsigned r = x.u + 0x7fffu + ((x.u >> 16) & 1u);
    return (u16)(r >> 16);
}

union FragU { v16bf v; uint4 q[2]; };

// A-matrix fragment (16x32 bf16), ISA layout: lane half selects K {0..7,16..23} / {8..15,24..31}
__device__ __forceinline__ v16bf load_fragA(const u16* p, int stride, int row_base, int k_base) {
    int lane = threadIdx.x & 31;
    const u16* q = p + (row_base + (lane & 15)) * stride + k_base + ((lane >> 4) << 3);
    FragU f;
    f.q[0] = *reinterpret_cast<const uint4*>(q);
    f.q[1] = *reinterpret_cast<const uint4*>(q + 16);
    return f.v;
}

// B-matrix fragment (32x16 bf16) loaded from B^T (row = N, contiguous along K):
// lanes 0-15 hold K={0..7,16..23}, lanes 16-31 hold K={8..15,24..31}
__device__ __forceinline__ v16bf load_fragB(const u16* p, int stride, int n_base, int k_base) {
    int lane = threadIdx.x & 31;
    const u16* q = p + (n_base + (lane & 15)) * stride + k_base + ((lane >> 4) << 3);
    FragU f;
    f.q[0] = *reinterpret_cast<const uint4*>(q);
    f.q[1] = *reinterpret_cast<const uint4*>(q + 16);
    return f.v;
}

#define WMMA_BF16(A, B, Cacc) \
    __builtin_amdgcn_wmma_f32_16x16x32_bf16(false, (A), false, (B), (short)0, (Cacc), false, false)

// ---------------------------------------------------------------------------
// Kernel 0: weight conversion (fp32 -> bf16, transposed) + padded bias table
// ---------------------------------------------------------------------------
__global__ __launch_bounds__(256) void prep_kernel(
    const float* __restrict__ qkv_w, const float* __restrict__ proj_w,
    const float* __restrict__ fc1_w, const float* __restrict__ fc2_w,
    const float* __restrict__ bias_table,
    u16* __restrict__ wqkvT, u16* __restrict__ wprojT,
    u16* __restrict__ wfc1T, u16* __restrict__ wfc2T,
    float* __restrict__ biasg)
{
    int t = blockIdx.x * 256 + threadIdx.x;
    if (t < 196608) {                     // qkv_w [256][768] -> [768][256]
        int n = t >> 8, k = t & 255;
        wqkvT[t] = f2bf(qkv_w[k * 768 + n]);
        return;
    }
    t -= 196608;
    if (t < 65536) {                      // proj_w [256][256] -> transposed
        int n = t >> 8, k = t & 255;
        wprojT[t] = f2bf(proj_w[k * 256 + n]);
        return;
    }
    t -= 65536;
    if (t < 262144) {                     // fc1_w [256][1024] -> [1024][256]
        int n = t >> 8, k = t & 255;
        wfc1T[t] = f2bf(fc1_w[k * 1024 + n]);
        return;
    }
    t -= 262144;
    if (t < 262144) {                     // fc2_w [1024][256] -> [256][1024]
        int n = t >> 10, k = t & 1023;
        wfc2T[t] = f2bf(fc2_w[k * 256 + n]);
        return;
    }
    t -= 262144;
    if (t < 32768) {                      // biasg [8][64][64], pad = -1e30 (mask)
        int head = t >> 12, rem = t & 4095, i = rem >> 6, j = rem & 63;
        float v;
        if (i < NT_ && j < NT_) {
            int yi = i / 7, xi = i % 7, yj = j / 7, xj = j % 7;
            int idx = (yi - yj + 6) * 13 + (xi - xj + 6);
            v = bias_table[idx * NH_ + head];
        } else {
            v = -1e30f;
        }
        biasg[t] = v;
    }
}

// ---------------------------------------------------------------------------
// Kernel 1: per-window fused LN1 + QKV + attention + proj + residual
// LDS map (244,736 B), all row strides bank-conflict-free:
//   [0      ,  33792) sQ  [64][264] bf16
//   [33792  ,  67584) sK  [64][264] bf16
//   [67584  , 104448) sV  [8][32][72] bf16  (v transposed: head, hd, token)
//   [104448 , 210944) stage1-2: sH [64][264] bf16 (first 33,792 B)
//                     stage3  : attnF [4][64][68] f32 (69,632 B)
//                               attnB [4][64][72] bf16 (36,864 B, at +69,632)
//   [210944 , 244736) sO  [64][264] bf16
// ---------------------------------------------------------------------------
__global__ __launch_bounds__(256) void attn_block_kernel(
    const float* __restrict__ x, const float* __restrict__ n1g, const float* __restrict__ n1b,
    const float* __restrict__ qkvb, const float* __restrict__ projb,
    const u16* __restrict__ wqkvT, const u16* __restrict__ wprojT,
    const float* __restrict__ biasg, float* __restrict__ out)
{
    __shared__ __align__(16) char smem[244736];
    u16*   sQ    = (u16*)smem;
    u16*   sK    = (u16*)(smem + 33792);
    u16*   sV    = (u16*)(smem + 67584);
    u16*   sH    = (u16*)(smem + 104448);
    float* attnF = (float*)(smem + 104448);
    u16*   attnB = (u16*)(smem + 104448 + 69632);
    u16*   sO    = (u16*)(smem + 210944);

    const int tid = threadIdx.x, lane = tid & 31, wave = tid >> 5;
    const int win = blockIdx.x;
    const int b = win >> 6, wh = (win >> 3) & 7, ww = win & 7;

    // warm L2 with proj weights while LN runs (global_prefetch_b8, no counters)
    __builtin_prefetch((const char*)wprojT + tid * 256, 0, 3);
    __builtin_prefetch((const char*)wprojT + 65536 + tid * 256, 0, 3);

    // ---- Stage 1: LayerNorm1 into sH (bf16), rows 49..63 zero-padded ----
    for (int r = wave; r < NP_; r += 8) {
        const int c0 = lane * 8;
        if (r < NT_) {
            const int gy = wh * 7 + r / 7, gx = ww * 7 + r % 7;
            const float* px = x + (((size_t)b * 56 + gy) * 56 + gx) * C_;
            float4 a0 = *(const float4*)(px + c0);
            float4 a1 = *(const float4*)(px + c0 + 4);
            float v0[8] = {a0.x, a0.y, a0.z, a0.w, a1.x, a1.y, a1.z, a1.w};
            float s = 0.f, ss = 0.f;
            #pragma unroll
            for (int e = 0; e < 8; e++) { s += v0[e]; ss += v0[e] * v0[e]; }
            #pragma unroll
            for (int m = 16; m > 0; m >>= 1) {
                s  += __shfl_xor(s,  m, 32);
                ss += __shfl_xor(ss, m, 32);
            }
            const float mean = s * (1.f / 256.f);
            const float var  = ss * (1.f / 256.f) - mean * mean;
            const float inv  = rsqrtf(var + 1e-5f);
            u16 o[8];
            #pragma unroll
            for (int e = 0; e < 8; e++) {
                const int c = c0 + e;
                o[e] = f2bf((v0[e] - mean) * inv * n1g[c] + n1b[c]);
            }
            uint4 pk;
            pk.x = (unsigned)o[0] | ((unsigned)o[1] << 16);
            pk.y = (unsigned)o[2] | ((unsigned)o[3] << 16);
            pk.z = (unsigned)o[4] | ((unsigned)o[5] << 16);
            pk.w = (unsigned)o[6] | ((unsigned)o[7] << 16);
            *(uint4*)(sH + r * SH_STR + c0) = pk;
        } else {
            uint4 z = {0u, 0u, 0u, 0u};
            *(uint4*)(sH + r * SH_STR + c0) = z;
        }
    }
    __syncthreads();

    // ---- Stage 2: QKV = h[64x256] @ Wqkv[256x768]; scatter into sQ/sK/sV ----
    for (int t = wave; t < 192; t += 8) {
        const int mt = t / 48, nt = t % 48;
        v8f acc = {};
        #pragma unroll
        for (int ks = 0; ks < 8; ks++) {
            v16bf a  = load_fragA(sH, SH_STR, mt * 16, ks * 32);
            v16bf bb = load_fragB(wqkvT, C_, nt * 16, ks * 32);
            acc = WMMA_BF16(a, bb, acc);
        }
        const int j = nt * 16 + (lane & 15);
        const float bias = qkvb[j];
        const int sec = j >> 8, head = (j >> 5) & 7, hd = j & 31;
        #pragma unroll
        for (int r = 0; r < 8; r++) {
            const int row = mt * 16 + r + ((lane >> 4) << 3);
            const float v = acc[r] + bias;
            if (sec == 0)      sQ[row * SH_STR + head * HD_ + hd] = f2bf(v * SCALE_);
            else if (sec == 1) sK[row * SH_STR + head * HD_ + hd] = f2bf(v);
            else               sV[head * (HD_ * SV_STR) + hd * SV_STR + row] = f2bf(v);
        }
    }
    __syncthreads();

    // ---- Stage 3: attention, 2 passes x 4 heads, 2 waves (32 rows) per head ----
    for (int p = 0; p < 2; p++) {
        const int head = p * 4 + (wave >> 1);
        const int M0   = (wave & 1) * 32;
        const int slot = wave >> 1;
        float* aF = attnF + slot * (NP_ * AF_STR);
        u16*   aB = attnB + slot * (NP_ * AB_STR);

        // Q*K^T (K=HD=32 -> single WMMA per tile) + bias
        #pragma unroll
        for (int mt = 0; mt < 2; mt++) {
            #pragma unroll
            for (int nt = 0; nt < 4; nt++) {
                v8f acc = {};
                v16bf a  = load_fragA(sQ + head * HD_, SH_STR, M0 + mt * 16, 0);
                v16bf bb = load_fragB(sK + head * HD_, SH_STR, nt * 16, 0);
                acc = WMMA_BF16(a, bb, acc);
                const int col = nt * 16 + (lane & 15);
                #pragma unroll
                for (int r = 0; r < 8; r++) {
                    const int row = M0 + mt * 16 + r + ((lane >> 4) << 3);
                    aF[row * AF_STR + col] = acc[r] + biasg[(head * NP_ + row) * NP_ + col];
                }
            }
        }
        // softmax (wave-local: each lane owns one of its 32 rows), vectorized
        {
            const int row = M0 + lane;
            float4* pr4 = (float4*)(aF + row * AF_STR);
            float mx = -3.4e38f;
            #pragma unroll
            for (int c = 0; c < 16; c++) {
                float4 v = pr4[c];
                mx = fmaxf(mx, fmaxf(fmaxf(v.x, v.y), fmaxf(v.z, v.w)));
            }
            float sum = 0.f;
            #pragma unroll
            for (int c = 0; c < 16; c++) {
                float4 v = pr4[c];
                v.x = __expf(v.x - mx); v.y = __expf(v.y - mx);
                v.z = __expf(v.z - mx); v.w = __expf(v.w - mx);
                sum += v.x + v.y + v.z + v.w;
                pr4[c] = v;
            }
            const float inv = 1.f / sum;
            u16* pw = aB + row * AB_STR;
            #pragma unroll
            for (int c = 0; c < 8; c++) {
                float4 v0 = pr4[2 * c], v1 = pr4[2 * c + 1];
                uint4 pk;
                pk.x = (unsigned)f2bf(v0.x * inv) | ((unsigned)f2bf(v0.y * inv) << 16);
                pk.y = (unsigned)f2bf(v0.z * inv) | ((unsigned)f2bf(v0.w * inv) << 16);
                pk.z = (unsigned)f2bf(v1.x * inv) | ((unsigned)f2bf(v1.y * inv) << 16);
                pk.w = (unsigned)f2bf(v1.z * inv) | ((unsigned)f2bf(v1.w * inv) << 16);
                *(uint4*)(pw + c * 8) = pk;
            }
        }
        // attn @ V  (K=64 -> 2 WMMA steps); write into sO
        #pragma unroll
        for (int mt = 0; mt < 2; mt++) {
            #pragma unroll
            for (int nt = 0; nt < 2; nt++) {
                v8f acc = {};
                #pragma unroll
                for (int ks = 0; ks < 2; ks++) {
                    v16bf a  = load_fragA(aB, AB_STR, M0 + mt * 16, ks * 32);
                    v16bf bb = load_fragB(sV + head * (HD_ * SV_STR), SV_STR, nt * 16, ks * 32);
                    acc = WMMA_BF16(a, bb, acc);
                }
                const int hd = nt * 16 + (lane & 15);
                #pragma unroll
                for (int r = 0; r < 8; r++) {
                    const int row = M0 + mt * 16 + r + ((lane >> 4) << 3);
                    sO[row * SH_STR + head * HD_ + hd] = f2bf(acc[r]);
                }
            }
        }
    }
    __syncthreads();

    // ---- Stage 4: proj + residual + window reverse (write x2 = x + o) ----
    for (int t = wave; t < 64; t += 8) {
        const int mt = t >> 4, nt = t & 15;
        v8f acc = {};
        #pragma unroll
        for (int ks = 0; ks < 8; ks++) {
            v16bf a  = load_fragA(sO, SH_STR, mt * 16, ks * 32);
            v16bf bb = load_fragB(wprojT, C_, nt * 16, ks * 32);
            acc = WMMA_BF16(a, bb, acc);
        }
        const int col = nt * 16 + (lane & 15);
        const float bias = projb[col];
        #pragma unroll
        for (int r = 0; r < 8; r++) {
            const int row = mt * 16 + r + ((lane >> 4) << 3);
            if (row < NT_) {
                const int gy = wh * 7 + row / 7, gx = ww * 7 + row % 7;
                const size_t oi = (((size_t)b * 56 + gy) * 56 + gx) * C_ + col;
                out[oi] = x[oi] + acc[r] + bias;
            }
        }
    }
}

// ---------------------------------------------------------------------------
// Kernel 2: fused LN2 + MLP (fc1 + GELU + fc2) + residual, 64 tokens/block
// LDS: sH [64][264] bf16 (33,792 B) + sM [64][1032] bf16 (132,096 B)
// ---------------------------------------------------------------------------
__global__ __launch_bounds__(256) void mlp_kernel(
    const float* __restrict__ n2g, const float* __restrict__ n2b,
    const float* __restrict__ fc1b, const float* __restrict__ fc2b,
    const u16* __restrict__ wfc1T, const u16* __restrict__ wfc2T,
    float* __restrict__ out)
{
    __shared__ __align__(16) char smem[165888];
    u16* sH = (u16*)smem;              // [64][264]
    u16* sM = (u16*)(smem + 33792);    // [64][1032]

    const int tid = threadIdx.x, lane = tid & 31, wave = tid >> 5;
    const size_t row0 = (size_t)blockIdx.x * 64;

    // warm L2 with fc2 weights while LN + fc1 run
    __builtin_prefetch((const char*)wfc2T + tid * 256, 0, 3);
    __builtin_prefetch((const char*)wfc2T + 65536 + tid * 256, 0, 3);

    // ---- LayerNorm2 on x2 (currently in `out`) ----
    for (int r = wave; r < 64; r += 8) {
        const int c0 = lane * 8;
        const float* px = out + (row0 + r) * C_;
        float4 a0 = *(const float4*)(px + c0);
        float4 a1 = *(const float4*)(px + c0 + 4);
        float v0[8] = {a0.x, a0.y, a0.z, a0.w, a1.x, a1.y, a1.z, a1.w};
        float s = 0.f, ss = 0.f;
        #pragma unroll
        for (int e = 0; e < 8; e++) { s += v0[e]; ss += v0[e] * v0[e]; }
        #pragma unroll
        for (int m = 16; m > 0; m >>= 1) {
            s  += __shfl_xor(s,  m, 32);
            ss += __shfl_xor(ss, m, 32);
        }
        const float mean = s * (1.f / 256.f);
        const float var  = ss * (1.f / 256.f) - mean * mean;
        const float inv  = rsqrtf(var + 1e-5f);
        u16 o[8];
        #pragma unroll
        for (int e = 0; e < 8; e++) {
            const int c = c0 + e;
            o[e] = f2bf((v0[e] - mean) * inv * n2g[c] + n2b[c]);
        }
        uint4 pk;
        pk.x = (unsigned)o[0] | ((unsigned)o[1] << 16);
        pk.y = (unsigned)o[2] | ((unsigned)o[3] << 16);
        pk.z = (unsigned)o[4] | ((unsigned)o[5] << 16);
        pk.w = (unsigned)o[6] | ((unsigned)o[7] << 16);
        *(uint4*)(sH + r * SH_STR + c0) = pk;
    }
    __syncthreads();

    // ---- fc1: [64x256] @ [256x1024] + bias, exact GELU -> sM bf16 ----
    for (int t = wave; t < 256; t += 8) {
        const int mt = t >> 6, nt = t & 63;
        v8f acc = {};
        #pragma unroll
        for (int ks = 0; ks < 8; ks++) {
            v16bf a  = load_fragA(sH, SH_STR, mt * 16, ks * 32);
            v16bf bb = load_fragB(wfc1T, C_, nt * 16, ks * 32);
            acc = WMMA_BF16(a, bb, acc);
        }
        const int col = nt * 16 + (lane & 15);
        const float bias = fc1b[col];
        #pragma unroll
        for (int r = 0; r < 8; r++) {
            const int row = mt * 16 + r + ((lane >> 4) << 3);
            const float v = acc[r] + bias;
            const float g = 0.5f * v * (1.f + erff(v * 0.70710678118654752f));
            sM[row * SM_STR + col] = f2bf(g);
        }
    }
    __syncthreads();

    // ---- fc2: [64x1024] @ [1024x256] + bias + residual into out ----
    for (int t = wave; t < 64; t += 8) {
        const int mt = t >> 4, nt = t & 15;
        v8f acc = {};
        #pragma unroll
        for (int ks = 0; ks < 32; ks++) {
            v16bf a  = load_fragA(sM, SM_STR, mt * 16, ks * 32);
            v16bf bb = load_fragB(wfc2T, 1024, nt * 16, ks * 32);
            acc = WMMA_BF16(a, bb, acc);
        }
        const int col = nt * 16 + (lane & 15);
        const float bias = fc2b[col];
        #pragma unroll
        for (int r = 0; r < 8; r++) {
            const int row = mt * 16 + r + ((lane >> 4) << 3);
            const size_t oi = (row0 + row) * C_ + col;
            out[oi] = out[oi] + acc[r] + bias;
        }
    }
}

// ---------------------------------------------------------------------------
extern "C" void kernel_launch(void* const* d_in, const int* in_sizes, int n_in,
                              void* d_out, int out_size, void* d_ws, size_t ws_size,
                              hipStream_t stream)
{
    const float* x      = (const float*)d_in[0];
    const float* n1g    = (const float*)d_in[1];
    const float* n1b    = (const float*)d_in[2];
    const float* qkv_w  = (const float*)d_in[3];
    const float* qkv_b  = (const float*)d_in[4];
    const float* btab   = (const float*)d_in[5];
    const float* proj_w = (const float*)d_in[6];
    const float* proj_b = (const float*)d_in[7];
    const float* n2g    = (const float*)d_in[8];
    const float* n2b    = (const float*)d_in[9];
    const float* fc1_w  = (const float*)d_in[10];
    const float* fc1_b  = (const float*)d_in[11];
    const float* fc2_w  = (const float*)d_in[12];
    const float* fc2_b  = (const float*)d_in[13];
    float* out = (float*)d_out;

    char* w = (char*)d_ws;
    u16*   wqkvT  = (u16*)(w);             // 768*256*2  = 393216
    u16*   wprojT = (u16*)(w + 393216);    // 256*256*2  = 131072
    u16*   wfc1T  = (u16*)(w + 524288);    // 1024*256*2 = 524288
    u16*   wfc2T  = (u16*)(w + 1048576);   // 256*1024*2 = 524288
    float* biasg  = (float*)(w + 1572864); // 8*64*64*4  = 131072  (total ~1.7 MB)

    prep_kernel<<<3200, 256, 0, stream>>>(qkv_w, proj_w, fc1_w, fc2_w, btab,
                                          wqkvT, wprojT, wfc1T, wfc2T, biasg);
    attn_block_kernel<<<2048, 256, 0, stream>>>(x, n1g, n1b, qkv_b, proj_b,
                                                wqkvT, wprojT, biasg, out);
    mlp_kernel<<<1568, 256, 0, stream>>>(n2g, n2b, fc1_b, fc2_b, wfc1T, wfc2T, out);
}
